// GNNModel_8340826489242
// MI455X (gfx1250) — matrix operand order repaired and checked by reading, not compiled, not run on hardware
//
#include <hip/hip_runtime.h>

// ---------------------------------------------------------------------------
// Types for CDNA5 WMMA (wave32, v_wmma_f32_16x16x32_bf16)
// ---------------------------------------------------------------------------
typedef __attribute__((ext_vector_type(16))) __bf16  v16bf;
typedef __attribute__((ext_vector_type(8)))  float   v8f;

// f32 -> bf16 RNE via the compiler's native lowering for __bf16
__device__ __forceinline__ v16bf cvt16(const float (&v)[16]) {
    v16bf r;
    #pragma unroll
    for (int i = 0; i < 16; ++i) r[i] = (__bf16)v[i];
    return r;
}

__device__ __forceinline__ float wred_sum(float v) {
    #pragma unroll
    for (int m = 16; m >= 1; m >>= 1) v += __shfl_xor(v, m, 32);
    return v;
}
__device__ __forceinline__ float wred_max(float v) {
    #pragma unroll
    for (int m = 16; m >= 1; m >>= 1) v = fmaxf(v, __shfl_xor(v, m, 32));
    return v;
}

// ---------------------------------------------------------------------------
// GEMM: C[M,N] = gather(A)[M,K] @ B[K,N] (+ bias[n]), bf16 WMMA, f32 accum.
// N,K are template constants -> B loads use immediate offsets, K-loop fully
// known. One wave computes a 16x32 tile (2 WMMAs sharing the A fragment);
// 4 waves/block. For N%32==0 the edge clamp/mask folds away at compile time.
// ---------------------------------------------------------------------------
#define GEMM_WAVES 4

template <int N, int K>
__global__ __launch_bounds__(128) void gemm_bf16_wmma(
    const float* __restrict__ A, const float* __restrict__ B,
    const float* __restrict__ bias, float* __restrict__ C,
    const int* __restrict__ gather, int M)
{
    const int wave   = threadIdx.x >> 5;
    const int lane   = threadIdx.x & 31;
    constexpr int npairs = (N + 31) / 32;       // 32-column pairs of tiles
    const int tm  = blockIdx.x;
    const int tnp = blockIdx.y * GEMM_WAVES + wave;
    if (npairs % GEMM_WAVES != 0) {             // constant-folded test
        if (tnp >= npairs) return;              // wave-uniform: EXEC all-ones
    }

    const int hi   = lane >> 4;                 // lane half: 0 or 1
    const int mloc = lane & 15;

    const int mrow = tm * 16 + mloc;
    int arow = (mrow < M) ? mrow : (M - 1);
    if (gather) arow = gather[arow];
    const float* __restrict__ Arow = A + (size_t)arow * K;

    const int n0 = tnp * 32 + mloc, n1 = n0 + 16;
    int   nc0 = n0, nc1 = n1;
    float msk0 = 1.0f, msk1 = 1.0f;
    if (N % 32 != 0) {                          // only the ragged-N instance
        nc0 = (n0 < N) ? n0 : (N - 1);
        nc1 = (n1 < N) ? n1 : (N - 1);
        msk0 = (n0 < N) ? 1.0f : 0.0f;
        msk1 = (n1 < N) ? 1.0f : 0.0f;
    }

    v8f acc0 = {}, acc1 = {};
    #pragma unroll 2
    for (int kb = 0; kb < K; kb += 32) {
        // ---- A fragment: k = kb + e + 16*hi → one contiguous run
        const int ka = kb + hi * 16;
        float4 a0 = *(const float4*)(Arow + ka);
        float4 a1 = *(const float4*)(Arow + ka + 4);
        float4 a2 = *(const float4*)(Arow + ka + 8);
        float4 a3 = *(const float4*)(Arow + ka + 12);
        const float av[16] = { a0.x, a0.y, a0.z, a0.w, a1.x, a1.y, a1.z, a1.w,
                               a2.x, a2.y, a2.z, a2.w, a3.x, a3.y, a3.z, a3.w };
        const v16bf af = cvt16(av);

        // ---- B fragments: k = kb + e + 16*hi; constant-stride offsets
        const float* __restrict__ Bp0 = B + (size_t)(kb + hi * 16) * N + nc0;
        const float* __restrict__ Bp1 = B + (size_t)(kb + hi * 16) * N + nc1;
        float b0[16], b1[16];
        #pragma unroll
        for (int e = 0; e < 16; ++e) {
            b0[e] = Bp0[e * N] * msk0;          // e*N is a compile-time imm
            b1[e] = Bp1[e * N] * msk1;
        }
        const v16bf bf0 = cvt16(b0);
        const v16bf bf1 = cvt16(b1);

        acc0 = __builtin_amdgcn_wmma_f32_16x16x32_bf16(
                   false, af, false, bf0, (short)0, acc0, false, false);
        acc1 = __builtin_amdgcn_wmma_f32_16x16x32_bf16(
                   false, af, false, bf1, (short)0, acc1, false, false);
    }

    const float bv0 = bias ? bias[nc0] : 0.0f;
    const float bv1 = bias ? bias[nc1] : 0.0f;
    #pragma unroll
    for (int r = 0; r < 8; ++r) {
        const int m = tm * 16 + r + hi * 8;
        if (m < M) {
            if (n0 < N) C[(size_t)m * N + n0] = acc0[r] + bv0;
            if (n1 < N) C[(size_t)m * N + n1] = acc1[r] + bv1;
        }
    }
}

template <int N, int K>
static inline void launch_gemm(const float* A, const float* B, const float* bias,
                               float* C, const int* gather, int M,
                               hipStream_t stream)
{
    constexpr int npairs = (N + 31) / 32;
    dim3 grid((M + 15) / 16, (npairs + GEMM_WAVES - 1) / GEMM_WAVES);
    gemm_bf16_wmma<N, K><<<grid, 128, 0, stream>>>(A, B, bias, C, gather, M);
}

// ---------------------------------------------------------------------------
// Per-node attention scores: a_s[n,h] = sum_c x[n,h*128+c]*attS[h,c] (wave/node)
// ---------------------------------------------------------------------------
__global__ __launch_bounds__(256) void attn_scores(
    const float* __restrict__ x, const float* __restrict__ attS,
    const float* __restrict__ attD, float* __restrict__ a_s,
    float* __restrict__ a_d, int Ns, int Nd)
{
    const int node = (blockIdx.x * blockDim.x + threadIdx.x) >> 5;
    const int lane = threadIdx.x & 31;
    if (node >= Ns) return;
    const float* __restrict__ xr = x + (size_t)node * 512;
    const bool dod = node < Nd;
    float ps[4] = {0, 0, 0, 0}, pd[4] = {0, 0, 0, 0};
    #pragma unroll
    for (int h = 0; h < 4; ++h)
        #pragma unroll
        for (int j = 0; j < 4; ++j) {
            const int c = h * 128 + lane + 32 * j;
            const float v = xr[c];
            ps[h] += v * attS[c];
            if (dod) pd[h] += v * attD[c];
        }
    #pragma unroll
    for (int h = 0; h < 4; ++h) {
        const float s = wred_sum(ps[h]);
        const float d = wred_sum(dod ? pd[h] : 0.0f);
        if (lane == 0) {
            a_s[node * 4 + h] = s;
            if (dod) a_d[node * 4 + h] = d;
        }
    }
}

// ---------------------------------------------------------------------------
// CSR row_ptr from sorted dst array (lower_bound per destination)
// ---------------------------------------------------------------------------
__global__ void build_rowptr(const int* __restrict__ dsts, int E, int n_dst,
                             int* __restrict__ rp)
{
    const int d = blockIdx.x * blockDim.x + threadIdx.x;
    if (d > n_dst) return;
    int lo = 0, hiE = E;
    while (lo < hiE) { const int mid = (lo + hiE) >> 1; if (dsts[mid] < d) lo = mid + 1; else hiE = mid; }
    rp[d] = lo;
}

// ---------------------------------------------------------------------------
// Segment softmax + weighted aggregation, one wave per dst node.
// o[d,c] = mean_h( sum_e coef[e,h]*x[src[e],h,c] ) + bias[c]
// ---------------------------------------------------------------------------
__global__ __launch_bounds__(256) void gat_aggregate(
    const float* __restrict__ x, const int* __restrict__ src,
    const float* __restrict__ a_s, const float* __restrict__ a_d,
    const int* __restrict__ rp, const float* __restrict__ bias,
    float* __restrict__ o, int n_dst)
{
    const int d    = (blockIdx.x * blockDim.x + threadIdx.x) >> 5;
    const int lane = threadIdx.x & 31;
    if (d >= n_dst) return;
    const int e0 = rp[d], e1 = rp[d + 1];

    float ad[4];
    #pragma unroll
    for (int h = 0; h < 4; ++h) ad[h] = a_d[d * 4 + h];

    // pass 1: per-head max logit
    float mh[4] = {-3.0e38f, -3.0e38f, -3.0e38f, -3.0e38f};
    for (int e = e0 + lane; e < e1; e += 32) {
        const int s = src[e];
        #pragma unroll
        for (int h = 0; h < 4; ++h) {
            float l = a_s[s * 4 + h] + ad[h];
            l = (l > 0.0f) ? l : 0.2f * l;
            mh[h] = fmaxf(mh[h], l);
        }
    }
    #pragma unroll
    for (int h = 0; h < 4; ++h) mh[h] = wred_max(mh[h]);

    // pass 2: per-head softmax denominator
    float sh[4] = {0, 0, 0, 0};
    for (int e = e0 + lane; e < e1; e += 32) {
        const int s = src[e];
        #pragma unroll
        for (int h = 0; h < 4; ++h) {
            float l = a_s[s * 4 + h] + ad[h];
            l = (l > 0.0f) ? l : 0.2f * l;
            sh[h] += __expf(l - mh[h]);
        }
    }
    #pragma unroll
    for (int h = 0; h < 4; ++h) sh[h] = fmaxf(wred_sum(sh[h]), 1e-16f);

    // pass 3: weighted accumulation; lane handles channels c = lane + 32*j
    float acc[4][4] = {};
    for (int e = e0; e < e1; ++e) {
        const int s = src[e];
        float w[4];
        #pragma unroll
        for (int h = 0; h < 4; ++h) {
            float l = a_s[s * 4 + h] + ad[h];
            l = (l > 0.0f) ? l : 0.2f * l;
            w[h] = __expf(l - mh[h]) / sh[h];
        }
        const float* __restrict__ xr = x + (size_t)s * 512;
        #pragma unroll
        for (int h = 0; h < 4; ++h)
            #pragma unroll
            for (int j = 0; j < 4; ++j)
                acc[h][j] += w[h] * xr[h * 128 + lane + 32 * j];
    }
    #pragma unroll
    for (int j = 0; j < 4; ++j) {
        const int c = lane + 32 * j;
        o[(size_t)d * 128 + c] =
            0.25f * (acc[0][j] + acc[1][j] + acc[2][j] + acc[3][j]) + bias[c];
    }
}

// ---------------------------------------------------------------------------
// BatchNorm helpers (C = 128 everywhere): stats via float atomics, then apply
// ---------------------------------------------------------------------------
__global__ void fzero(float* __restrict__ p, int n) {
    const int i = blockIdx.x * blockDim.x + threadIdx.x;
    if (i < n) p[i] = 0.0f;
}

__global__ __launch_bounds__(128) void bn_stats(
    const float* __restrict__ X, int M, float* __restrict__ stats)
{
    const int c   = threadIdx.x;                // column 0..127
    const int rpb = (M + gridDim.x - 1) / gridDim.x;
    const int r0  = blockIdx.x * rpb;
    int r1 = r0 + rpb; if (r1 > M) r1 = M;
    float s = 0.0f, ss = 0.0f;
    for (int r = r0; r < r1; ++r) {
        const float v = X[(size_t)r * 128 + c];
        s += v; ss += v * v;
    }
    atomicAdd(&stats[c], s);
    atomicAdd(&stats[128 + c], ss);
}

// act: 0=none, 1=elu, 2=relu. Writes rows [0,rows) to dst[r*stride + off + c]
__global__ __launch_bounds__(256) void bn_apply(
    const float* __restrict__ X, const float* __restrict__ stats,
    const float* __restrict__ g, const float* __restrict__ b,
    float* __restrict__ dst, int rows, int M, int dstStride, int dstOff, int act)
{
    const int i = blockIdx.x * blockDim.x + threadIdx.x;
    if (i >= rows * 128) return;
    const int r = i >> 7, c = i & 127;
    const float invM = 1.0f / (float)M;
    const float mu   = stats[c] * invM;
    const float var  = stats[128 + c] * invM - mu * mu;
    float y = (X[(size_t)r * 128 + c] - mu) * rsqrtf(var + 1e-5f) * g[c] + b[c];
    if (act == 1)      y = (y > 0.0f) ? y : (__expf(y) - 1.0f);
    else if (act == 2) y = fmaxf(y, 0.0f);
    dst[(size_t)r * dstStride + dstOff + c] = y;
}

// ---------------------------------------------------------------------------
// Attention pooling over P=6 branches: softmax(temp@wp+bp) weighted sum.
// One wave per batch row.
// ---------------------------------------------------------------------------
__global__ __launch_bounds__(256) void pool_attn(
    const float* __restrict__ temp, const float* __restrict__ wp,
    const float* __restrict__ bp, float* __restrict__ out, int batch)
{
    const int bidx = (blockIdx.x * blockDim.x + threadIdx.x) >> 5;
    const int lane = threadIdx.x & 31;
    if (bidx >= batch) return;
    const float* __restrict__ t = temp + (size_t)bidx * 768;
    float sc[6];
    #pragma unroll
    for (int p = 0; p < 6; ++p) {
        float part = 0.0f;
        #pragma unroll
        for (int j = 0; j < 4; ++j) {
            const int c = lane + 32 * j;
            part += t[p * 128 + c] * wp[c];
        }
        sc[p] = wred_sum(part) + bp[0];
    }
    float mx = sc[0];
    #pragma unroll
    for (int p = 1; p < 6; ++p) mx = fmaxf(mx, sc[p]);
    float den = 0.0f;
    #pragma unroll
    for (int p = 0; p < 6; ++p) { sc[p] = __expf(sc[p] - mx); den += sc[p]; }
    const float inv = 1.0f / den;
    #pragma unroll
    for (int j = 0; j < 4; ++j) {
        const int c = lane + 32 * j;
        float v = 0.0f;
        #pragma unroll
        for (int p = 0; p < 6; ++p) v += sc[p] * inv * t[p * 128 + c];
        out[(size_t)bidx * 128 + c] = v;
    }
}

// ---------------------------------------------------------------------------
// Host orchestration
// ---------------------------------------------------------------------------
extern "C" void kernel_launch(void* const* d_in, const int* in_sizes, int n_in,
                              void* d_out, int out_size, void* d_ws, size_t ws_size,
                              hipStream_t stream)
{
    (void)in_sizes; (void)n_in; (void)out_size; (void)ws_size;
    static const int N_TOTAL = 100000, ET = 5;
    static const int N_SRC_[2] = {40000, 20000}, N_DST_[2] = {30000, 10000};
    static const int BATCH_[2] = {25000, 8000}, E_[2] = {200000, 80000};

    // ---- inputs (setup_inputs order) ----
    const float* feature = (const float*)d_in[0];
    const int*   nidx[2] = {(const int*)d_in[1], (const int*)d_in[4]};
    const int*   esrc[2] = {(const int*)d_in[2], (const int*)d_in[5]};
    const int*   edst[2] = {(const int*)d_in[3], (const int*)d_in[6]};
    const float *Wsk[2], *bsk[2], *bng[2], *bnb[2], *Wg[2], *attS[2], *attD[2],
                *bg[2], *wp[2], *bp[2], *pg[2], *pb[2];
    for (int l = 0; l < 2; ++l) {
        const int base = 7 + l * 12;
        Wsk[l]  = (const float*)d_in[base + 0];  bsk[l] = (const float*)d_in[base + 1];
        bng[l]  = (const float*)d_in[base + 2];  bnb[l] = (const float*)d_in[base + 3];
        Wg[l]   = (const float*)d_in[base + 4];  attS[l] = (const float*)d_in[base + 5];
        attD[l] = (const float*)d_in[base + 6];  bg[l]  = (const float*)d_in[base + 7];
        wp[l]   = (const float*)d_in[base + 8];  bp[l]  = (const float*)d_in[base + 9];
        pg[l]   = (const float*)d_in[base + 10]; pb[l]  = (const float*)d_in[base + 11];
    }
    const float* Wm1 = (const float*)d_in[31];
    const float* bm1 = (const float*)d_in[32];
    const float* mg  = (const float*)d_in[33];
    const float* mb  = (const float*)d_in[34];
    const float* Wm2 = (const float*)d_in[35];
    const float* bm2 = (const float*)d_in[36];

    // ---- workspace layout (aliased so the two big phases share R0) ----
    char* ws = (char*)d_ws;
    size_t off = 0;
    auto walloc = [&](size_t bytes) -> void* {
        void* p = ws + off; off = (off + bytes + 255) & ~(size_t)255; return p;
    };
    float* R0 = (float*)walloc((size_t)20480000 * 4); // skip-gemm out / gat x / mlp h
    float* R1 = (float*)walloc((size_t)3840000 * 4);  // gat aggregate out (o)
    float* R2 = (float*)walloc((size_t)19200000 * 4); // temp [batch,6,128] (both layers)
    float* R3 = (float*)walloc((size_t)3200000 * 4);  // f0
    float* R4 = (float*)walloc((size_t)1024000 * 4);  // f1
    float* AS = (float*)walloc((size_t)160000 * 4);   // a_s
    float* AD = (float*)walloc((size_t)120000 * 4);   // a_d
    int*   RP = (int*)  walloc((size_t)30016 * 4);    // row_ptr
    float* ST = (float*)walloc((size_t)256 * 4);      // bn stats: sum[128], sumsq[128]

    auto stats = [&](const float* X, int M) {
        fzero<<<1, 256, 0, stream>>>(ST, 256);
        bn_stats<<<256, 128, 0, stream>>>(X, M, ST);
    };
    auto apply = [&](const float* X, const float* g, const float* b, float* dst,
                     int rows, int M, int dstStride, int dstOff, int act) {
        const int n = rows * 128;
        bn_apply<<<(n + 255) / 256, 256, 0, stream>>>(X, ST, g, b, dst, rows, M,
                                                      dstStride, dstOff, act);
    };

    auto run_layer = [&](const float* feat, int li, float* fout) {
        const int Ns = N_SRC_[li], Nd = N_DST_[li], batch = BATCH_[li], E = E_[li];
        const int Mskip = (li == 0) ? N_TOTAL : BATCH_[0];

        // skip branch: elu(bn(feat @ Wsk + bsk)) -> temp slot 0
        if (li == 0) launch_gemm<128, 512>(feat, Wsk[li], bsk[li], R0, nullptr, Mskip, stream);
        else         launch_gemm<128, 128>(feat, Wsk[li], bsk[li], R0, nullptr, Mskip, stream);
        stats(R0, Mskip);
        apply(R0, bng[li] + 0, bnb[li] + 0, R2, batch, Mskip, 768, 0, 1);

        for (int e2 = 0; e2 < ET; ++e2) {
            // x = feat[nodes_idx[e2]] @ Wg[e2]  (gather fused into GEMM)
            const float* W = Wg[li] + (size_t)e2 * (li == 0 ? 512 : 128) * 512;
            const int* gi = nidx[li] + (size_t)e2 * Ns;
            if (li == 0) launch_gemm<512, 512>(feat, W, nullptr, R0, gi, Ns, stream);
            else         launch_gemm<512, 128>(feat, W, nullptr, R0, gi, Ns, stream);
            attn_scores<<<((size_t)Ns * 32 + 255) / 256, 256, 0, stream>>>(
                R0, attS[li] + (size_t)e2 * 512, attD[li] + (size_t)e2 * 512,
                AS, AD, Ns, Nd);
            build_rowptr<<<(Nd + 1 + 255) / 256, 256, 0, stream>>>(
                edst[li] + (size_t)e2 * E, E, Nd, RP);
            gat_aggregate<<<((size_t)Nd * 32 + 255) / 256, 256, 0, stream>>>(
                R0, esrc[li] + (size_t)e2 * E, AS, AD, RP,
                bg[li] + (size_t)e2 * 128, R1, Nd);
            stats(R1, Nd);
            apply(R1, bng[li] + (e2 + 1) * 128, bnb[li] + (e2 + 1) * 128, R2,
                  batch, Nd, 768, (e2 + 1) * 128, 1);
        }
        // attention pooling across the 6 branches, then bn(sk, pg, pb)
        pool_attn<<<((size_t)batch * 32 + 255) / 256, 256, 0, stream>>>(
            R2, wp[li], bp[li], fout, batch);
        stats(fout, batch);
        apply(fout, pg[li], pb[li], fout, batch, batch, 128, 0, 0);
    };

    run_layer(feature, 0, R3);   // f0 [25000,128]
    run_layer(R3, 1, R4);        // f1 [8000,128]

    // MLP head: relu(bn(f1 @ Wm1 + bm1, mg, mb)) @ Wm2 + bm2
    launch_gemm<128, 128>(R4, Wm1, bm1, R0, nullptr, 8000, stream);
    stats(R0, 8000);
    apply(R0, mg, mb, R0, 8000, 8000, 128, 0, 2);
    launch_gemm<153, 128>(R0, Wm2, bm2, (float*)d_out, nullptr, 8000, stream);
}
